// GraphUnetV5_40733469835306
// MI455X (gfx1250) — matrix operand order repaired and checked by reading, not compiled, not run on hardware
//
#include <hip/hip_runtime.h>
#include <hip/hip_bf16.h>
#include <math.h>

// ===================== CDNA5 / gfx1250 =====================
// wave32; matrix ops via V_WMMA_F32_16X16X32_BF16 (f32 accumulate).
// Weight tiles staged with GLOBAL_LOAD_ASYNC_TO_LDS_B128 (ASYNCcnt).
typedef __attribute__((ext_vector_type(16))) __bf16 bf16x16;
typedef __attribute__((ext_vector_type(8)))  float  f32x8;
typedef __bf16 bf16;

#define TM 64
#define TN 128
#define TK 32
#define NTHR 128   // 4 waves per workgroup; wave tile = 32x64 (2x4 WMMA accs)

// ---------- WMMA fragment loaders (ISA 7.12.2 layouts) ----------
__device__ __forceinline__ bf16x16 fragA(const bf16* row, int half) {
  union { bf16x16 v; uint4 q[2]; } u;
  u.q[0] = *(const uint4*)(row + half * 8);
  u.q[1] = *(const uint4*)(row + half * 8 + 16);
  return u.v;
}
__device__ __forceinline__ bf16x16 fragB(const bf16* col, int half) {
  union { bf16x16 v; uint4 q[2]; } u;
  u.q[0] = *(const uint4*)(col + half * 8);
  u.q[1] = *(const uint4*)(col + half * 8 + 16);
  return u.v;
}
union bf4u { bf16 h[4]; uint2 u2; };
__device__ __forceinline__ void store_bf4(bf16* dst, float4 v) {
  bf4u t;
  t.h[0] = (bf16)v.x; t.h[1] = (bf16)v.y; t.h[2] = (bf16)v.z; t.h[3] = (bf16)v.w;
  *(uint2*)dst = t.u2;
}
#define WMMA_BF16(a, b, c) \
  __builtin_amdgcn_wmma_f32_16x16x32_bf16(false, (a), false, (b), (short)0, (c), false, false)

// ---------- CDNA5 async copy: global bf16 -> LDS, 16B per lane ----------
// the LDS destination address comes from the VDST VGPR; flat shared pointers carry the LDS byte offset in
// their low 32 bits, so truncation yields the VDST value.
__device__ __forceinline__ void async_ld_b128(void* lds, const void* gaddr) {
  unsigned ldsOff = (unsigned)(unsigned long long)lds;
  unsigned long long ga = (unsigned long long)gaddr;
  asm volatile("global_load_async_to_lds_b128 %0, %1, off"
               :: "v"(ldsOff), "v"(ga) : "memory");
}
__device__ __forceinline__ void wait_async0() {
  asm volatile("s_wait_asynccnt 0x0" ::: "memory");
}

// ===================== generic WMMA GEMM =====================
// C[z] = act(scale * A[z]*B[z] + bias) (+= C if accum)
// A is either pre-cast bf16 (Ah, weights; async-staged, never transposed) or
// f32 (Af, activations; vector/transA staging). z = (batch, head) with strides.
struct GemmArgs {
  const bf16* Ah; const float* Af; const float* Bm; float* C; const float* bias;
  long lda, ldb, ldc;
  long sAb, sAh, sBb, sBh, sCb, sCh;
  int M, N, K, zDiv, transA, transB, act, accum;
  float scale;
};

__global__ __launch_bounds__(NTHR) void k_gemm_wmma(GemmArgs g) {
  __shared__ __align__(16) bf16 sA[2][TM][TK + 8];
  __shared__ __align__(16) bf16 sB[2][TN][TK + 8];
  const int tid = threadIdx.x;
  const int wave = tid >> 5, lane = tid & 31;
  const int half = lane >> 4, r = lane & 15;
  const int z  = blockIdx.z;
  const int zb = z / g.zDiv, zh = z % g.zDiv;
  const bf16*  Abf = g.Ah ? g.Ah + (long)zb * g.sAb + (long)zh * g.sAh : nullptr;
  const float* Af  = g.Af ? g.Af + (long)zb * g.sAb + (long)zh * g.sAh : nullptr;
  const float* Bp  = g.Bm + (long)zb * g.sBb + (long)zh * g.sBh;
  float*       C   = g.C  + (long)zb * g.sCb + (long)zh * g.sCh;
  const int m0 = blockIdx.y * TM, n0 = blockIdx.x * TN;
  const int wm = (wave >> 1) * 32, wn = (wave & 1) * 64;
  const bool fullM = (m0 + TM <= g.M), fullN = (n0 + TN <= g.N);
  f32x8 acc[2][4] = {};

  auto stage = [&](int buf, int kk) {
    const bool fullK = (kk + TK <= g.K);
    // ---------------- A ----------------
    if (Abf) {
      if (fullM && fullK) {               // async bf16 tile copy (2 x b128/lane)
        for (int ci = tid; ci < TM * TK / 8; ci += NTHR) {
          int m = ci >> 2, kc = (ci & 3) << 3;
          async_ld_b128(&sA[buf][m][kc], Abf + (long)(m0 + m) * g.lda + kk + kc);
        }
      } else {
        for (int i = tid; i < TM * TK; i += NTHR) {
          int m = i >> 5, k = i & 31;
          int gm = m0 + m, gk = kk + k;
          bf16 v = (bf16)0.f;
          if (gm < g.M && gk < g.K) v = Abf[(long)gm * g.lda + gk];
          sA[buf][m][k] = v;
        }
      }
    } else if (fullM && fullK) {          // f32 activation A, vectorized
      if (!g.transA) {
        for (int gi = tid; gi < TM * TK / 4; gi += NTHR) {
          int m = gi >> 3, k = (gi & 7) << 2;
          float4 v = *(const float4*)(Af + (long)(m0 + m) * g.lda + kk + k);
          store_bf4(&sA[buf][m][k], v);
        }
      } else {
        for (int gi = tid; gi < TM * TK / 4; gi += NTHR) {
          int k = gi >> 4, m = (gi & 15) << 2;
          float4 v = *(const float4*)(Af + (long)(kk + k) * g.lda + m0 + m);
          sA[buf][m][k]     = (bf16)v.x; sA[buf][m + 1][k] = (bf16)v.y;
          sA[buf][m + 2][k] = (bf16)v.z; sA[buf][m + 3][k] = (bf16)v.w;
        }
      }
    } else {
      for (int i = tid; i < TM * TK; i += NTHR) {
        int m = i >> 5, k = i & 31;
        int gm = m0 + m, gk = kk + k;
        float v = 0.f;
        if (gm < g.M && gk < g.K)
          v = g.transA ? Af[(long)gk * g.lda + gm] : Af[(long)gm * g.lda + gk];
        sA[buf][m][k] = (bf16)v;
      }
    }
    // ---------------- B (f32 -> bf16, transposed into [n][k]) ----------------
    if (fullN && fullK) {
      if (!g.transB) {
        for (int gi = tid; gi < TN * TK / 4; gi += NTHR) {
          int k = gi >> 5, n = (gi & 31) << 2;
          float4 v = *(const float4*)(Bp + (long)(kk + k) * g.ldb + n0 + n);
          sB[buf][n][k]     = (bf16)v.x; sB[buf][n + 1][k] = (bf16)v.y;
          sB[buf][n + 2][k] = (bf16)v.z; sB[buf][n + 3][k] = (bf16)v.w;
        }
      } else {
        for (int gi = tid; gi < TN * TK / 4; gi += NTHR) {
          int n = gi >> 3, k = (gi & 7) << 2;
          float4 v = *(const float4*)(Bp + (long)(n0 + n) * g.ldb + kk + k);
          store_bf4(&sB[buf][n][k], v);
        }
      }
    } else {
      for (int i = tid; i < TN * TK; i += NTHR) {
        int n = i & 127, k = i >> 7;
        int gn = n0 + n, gk = kk + k;
        float v = 0.f;
        if (gn < g.N && gk < g.K)
          v = g.transB ? Bp[(long)gn * g.ldb + gk] : Bp[(long)gk * g.ldb + gn];
        sB[buf][n][k] = (bf16)v;
      }
    }
  };

  stage(0, 0);
  int cur = 0;
  for (int k0 = 0; k0 < g.K; k0 += TK) {
    wait_async0();                        // my async stores visible before signal
    __syncthreads();
    if (k0 + TK < g.K) {
      if (!Abf && !g.transA && k0 + 2 * TK < g.K)
        __builtin_prefetch(Af + (long)(m0 + (tid >> 1)) * g.lda + (k0 + 2 * TK) + (tid & 1) * 16);
      stage(cur ^ 1, k0 + TK);            // overlap staging with WMMA below
    }
    bf16x16 a0 = fragA(&sA[cur][wm + r][0], half);
    bf16x16 a1 = fragA(&sA[cur][wm + 16 + r][0], half);
    bf16x16 b0 = fragB(&sB[cur][wn + r][0], half);
    bf16x16 b1 = fragB(&sB[cur][wn + 16 + r][0], half);
    bf16x16 b2 = fragB(&sB[cur][wn + 32 + r][0], half);
    bf16x16 b3 = fragB(&sB[cur][wn + 48 + r][0], half);
    acc[0][0] = WMMA_BF16(a0, b0, acc[0][0]);
    acc[0][1] = WMMA_BF16(a0, b1, acc[0][1]);
    acc[0][2] = WMMA_BF16(a0, b2, acc[0][2]);
    acc[0][3] = WMMA_BF16(a0, b3, acc[0][3]);
    acc[1][0] = WMMA_BF16(a1, b0, acc[1][0]);
    acc[1][1] = WMMA_BF16(a1, b1, acc[1][1]);
    acc[1][2] = WMMA_BF16(a1, b2, acc[1][2]);
    acc[1][3] = WMMA_BF16(a1, b3, acc[1][3]);
    cur ^= 1;
  }
  for (int i = 0; i < 2; ++i) {
    int gmBase = m0 + wm + i * 16 + half * 8;
    for (int j = 0; j < 4; ++j) {
      int gn = n0 + wn + j * 16 + r;
      if (gn >= g.N) continue;
      for (int e = 0; e < 8; ++e) {
        int gm = gmBase + e;
        if (gm >= g.M) continue;
        float c = acc[i][j][e] * g.scale;
        if (g.bias) c += g.bias[gm];
        if (g.act == 1) c = c > 0.f ? c : 0.f;
        long off = (long)gm * g.ldc + gn;
        if (g.accum) c += C[off];
        C[off] = c;
      }
    }
  }
}

// ===================== implicit-GEMM 3x3 conv (pad 1) =====================
// Y[z][co][p] = act(W[co][:]*im2col(X[z])[:,p] + bias[co]), K = Cin*9.
struct ConvArgs {
  const bf16* Wt;    // [Cout][Cin*9] pre-cast bf16 (OIHW flat)
  const float* X;    // [B][Cin][H][W]
  float* Y;
  const float* bias;
  long sXz, sYz;
  int Cin, Cout, H, Wd, act;
};

__global__ __launch_bounds__(NTHR) void k_conv3_wmma(ConvArgs g) {
  __shared__ __align__(16) bf16 sA[2][TM][TK + 8];
  __shared__ __align__(16) bf16 sB[2][TN][TK + 8];
  const int tid = threadIdx.x;
  const int wave = tid >> 5, lane = tid & 31;
  const int half = lane >> 4, r = lane & 15;
  const int HW = g.H * g.Wd;
  const int K = g.Cin * 9;
  const float* X = g.X + (long)blockIdx.z * g.sXz;
  float*       Y = g.Y + (long)blockIdx.z * g.sYz;
  const int m0 = blockIdx.y * TM, n0 = blockIdx.x * TN;
  const int wm = (wave >> 1) * 32, wn = (wave & 1) * 64;
  const bool fullM = (m0 + TM <= g.Cout), fullN = (n0 + TN <= HW);
  f32x8 acc[2][4] = {};

  auto stage = [&](int buf, int kk) {
    const bool fullK = (kk + TK <= K);
    if (fullM && fullK) {                 // async weight tile copy
      for (int ci = tid; ci < TM * TK / 8; ci += NTHR) {
        int m = ci >> 2, kc = (ci & 3) << 3;
        async_ld_b128(&sA[buf][m][kc], g.Wt + (long)(m0 + m) * K + kk + kc);
      }
    } else {
      for (int i = tid; i < TM * TK; i += NTHR) {
        int m = i >> 5, k = i & 31;
        int gm = m0 + m, gk = kk + k;
        bf16 v = (bf16)0.f;
        if (gm < g.Cout && gk < K) v = g.Wt[(long)gm * K + gk];
        sA[buf][m][k] = v;
      }
    }
    if (fullN && fullK) {                 // im2col fast path: 4 pixels/iter
      for (int gi = tid; gi < TN * TK / 4; gi += NTHR) {
        int k = gi >> 5, n = (gi & 31) << 2;
        int gk = kk + k, p = n0 + n;
        int c = gk / 9, t = gk - 9 * c;
        int dh = t / 3 - 1, dw = t - (t / 3) * 3 - 1;
        int row = p / g.Wd, col = p - row * g.Wd;
        int h = row + dh;
        if (h >= 0 && h < g.H && col + dw >= 0 && col + 3 + dw < g.Wd && col + 3 < g.Wd) {
          float4 v = *(const float4*)(X + ((long)c * g.H + h) * g.Wd + col + dw);
          sB[buf][n][k]     = (bf16)v.x; sB[buf][n + 1][k] = (bf16)v.y;
          sB[buf][n + 2][k] = (bf16)v.z; sB[buf][n + 3][k] = (bf16)v.w;
        } else {
          for (int q = 0; q < 4; ++q) {
            int pq = p + q;
            int rq = pq / g.Wd, cq = pq - rq * g.Wd;
            int hh = rq + dh, ww = cq + dw;
            float v = 0.f;
            if (hh >= 0 && hh < g.H && ww >= 0 && ww < g.Wd)
              v = X[((long)c * g.H + hh) * g.Wd + ww];
            sB[buf][n + q][k] = (bf16)v;
          }
        }
      }
    } else {
      for (int i = tid; i < TN * TK; i += NTHR) {
        int n = i & 127, k = i >> 7;
        int gn = n0 + n, gk = kk + k;
        float v = 0.f;
        if (gn < HW && gk < K) {
          int c = gk / 9, t = gk - 9 * c;
          int h = gn / g.Wd + (t / 3) - 1;
          int w = gn - (gn / g.Wd) * g.Wd + (t % 3) - 1;
          if (h >= 0 && h < g.H && w >= 0 && w < g.Wd)
            v = X[((long)c * g.H + h) * g.Wd + w];
        }
        sB[buf][n][k] = (bf16)v;
      }
    }
  };

  stage(0, 0);
  int cur = 0;
  for (int k0 = 0; k0 < K; k0 += TK) {
    wait_async0();
    __syncthreads();
    if (k0 + TK < K) stage(cur ^ 1, k0 + TK);
    bf16x16 a0 = fragA(&sA[cur][wm + r][0], half);
    bf16x16 a1 = fragA(&sA[cur][wm + 16 + r][0], half);
    bf16x16 b0 = fragB(&sB[cur][wn + r][0], half);
    bf16x16 b1 = fragB(&sB[cur][wn + 16 + r][0], half);
    bf16x16 b2 = fragB(&sB[cur][wn + 32 + r][0], half);
    bf16x16 b3 = fragB(&sB[cur][wn + 48 + r][0], half);
    acc[0][0] = WMMA_BF16(a0, b0, acc[0][0]);
    acc[0][1] = WMMA_BF16(a0, b1, acc[0][1]);
    acc[0][2] = WMMA_BF16(a0, b2, acc[0][2]);
    acc[0][3] = WMMA_BF16(a0, b3, acc[0][3]);
    acc[1][0] = WMMA_BF16(a1, b0, acc[1][0]);
    acc[1][1] = WMMA_BF16(a1, b1, acc[1][1]);
    acc[1][2] = WMMA_BF16(a1, b2, acc[1][2]);
    acc[1][3] = WMMA_BF16(a1, b3, acc[1][3]);
    cur ^= 1;
  }
  for (int i = 0; i < 2; ++i) {
    int gmBase = m0 + wm + i * 16 + half * 8;
    for (int j = 0; j < 4; ++j) {
      int gn = n0 + wn + j * 16 + r;
      if (gn >= HW) continue;
      for (int e = 0; e < 8; ++e) {
        int gm = gmBase + e;
        if (gm >= g.Cout) continue;
        float c = acc[i][j][e];
        if (g.bias) c += g.bias[gm];
        if (g.act == 1) c = c > 0.f ? c : 0.f;
        Y[(long)gm * HW + gn] = c;
      }
    }
  }
}

// ===================== support kernels =====================
#define GS(i, total) for (long i = blockIdx.x * (long)blockDim.x + threadIdx.x; \
                          i < (total); i += (long)gridDim.x * blockDim.x)

__global__ void k_cast(const float* x, bf16* y, long n) { GS(i, n) y[i] = (bf16)x[i]; }

__global__ void k_bn_stats(const float* x, float* mean, float* rstd,
                           long chStride, long zStride, int Bc, int HW) {
  const int c = blockIdx.x, tid = threadIdx.x;
  float s = 0.f, s2 = 0.f;
  for (int b = 0; b < Bc; ++b) {
    const float* p = x + (long)b * zStride + (long)c * chStride;
    for (int i = tid; i < HW; i += blockDim.x) { float v = p[i]; s += v; s2 += v * v; }
  }
  __shared__ float rs[256], rq[256];
  rs[tid] = s; rq[tid] = s2; __syncthreads();
  for (int st = 128; st > 0; st >>= 1) {
    if (tid < st) { rs[tid] += rs[tid + st]; rq[tid] += rq[tid + st]; }
    __syncthreads();
  }
  if (tid == 0) {
    float n = (float)Bc * (float)HW;
    float m = rs[0] / n;
    float v = rq[0] / n - m * m;
    mean[c] = m; rstd[c] = rsqrtf(v + 1e-5f);
  }
}

__global__ void k_bn_apply(float* x, const float* mean, const float* rstd,
                           int C, long chStride, long zStride, int HW, int act, long total) {
  GS(i, total) {
    int hw = (int)(i % HW); long t = i / HW;
    int c = (int)(t % C); int b = (int)(t / C);
    long off = (long)b * zStride + (long)c * chStride + hw;
    float v = (x[off] - mean[c]) * rstd[c];
    if (act == 1) v = v > 0.f ? v : 0.f;
    else if (act == 2) v = v > 0.f ? v : 0.01f * v;
    x[off] = v;
  }
}

__global__ void k_pool2(const float* x, float* y, int C, int H, int W, long total) {
  int Ho = H >> 1, Wo = W >> 1;
  GS(i, total) {
    int w = (int)(i % Wo); long t = i / Wo;
    int h = (int)(t % Ho); t /= Ho;
    int c = (int)(t % C);  int b = (int)(t / C);
    const float* p = x + (((long)b * C + c) * H + 2 * h) * W + 2 * w;
    float m = fmaxf(fmaxf(p[0], p[1]), fmaxf(p[W], p[W + 1]));
    y[(((long)b * C + c) * Ho + h) * Wo + w] = m;
  }
}

__global__ void k_pack_upw(const float* w, bf16* wp, int Ci, int Co, long total) {
  GS(i, total) {  // wp[sub][o][c] = w[c][o][dh][dw], bf16 out
    int c = (int)(i % Ci); long t = i / Ci;
    int o = (int)(t % Co); int sub = (int)(t / Co);
    wp[i] = (bf16)w[(long)c * Co * 4 + (long)o * 4 + sub];
  }
}

__global__ void k_up_inter(const float* t, const float* bias, float* y,
                           int Co, int Hin, int Win, long sTz, long sYz, long total) {
  GS(i, total) {
    long hw = i % ((long)Hin * Win); long r = i / ((long)Hin * Win);
    int o = (int)(r % Co); long r2 = r / Co;
    int sub = (int)(r2 % 4); int b = (int)(r2 / 4);
    int dh = sub >> 1, dw = sub & 1;
    int h = (int)(hw / Win), w = (int)(hw % Win);
    int Wo = 2 * Win;
    y[(long)b * sYz + (long)o * 4 * Hin * Win + (long)(2 * h + dh) * Wo + (2 * w + dw)]
      = t[(long)b * sTz + ((long)sub * Co + o) * Hin * Win + hw] + bias[o];
  }
}

__global__ void k_copy_ch(const float* src, float* dst, int C, long L,
                          long sSz, long sDz, long total) {
  GS(i, total) {
    long l = i % L; long r = i / L;
    int c = (int)(r % C); int b = (int)(r / C);
    dst[(long)b * sDz + (long)c * L + l] = src[(long)b * sSz + (long)c * L + l];
  }
}

__global__ void k_add(float* out, const float* a, const float* b, long total) {
  GS(i, total) out[i] = a[i] + b[i];
}

__global__ void k_zero(float* p, long total) { GS(i, total) p[i] = 0.f; }

__global__ void k_gather(const float* feat, const int* kp, float* gf,
                         int C, int Nk, int Wf, int shift, long sFz, long chS, long total) {
  GS(i, total) {
    int n = (int)(i % Nk); long r = i / Nk;
    int c = (int)(r % C); int b = (int)(r / C);
    int r0 = kp[((long)b * Nk + n) * 2] >> shift;
    int c0 = kp[((long)b * Nk + n) * 2 + 1] >> shift;
    long idx = (long)r0 * Wf + c0;
    gf[((long)b * C + c) * Nk + n] = feat[(long)b * sFz + (long)c * chS + idx];
  }
}

// deterministic scatter: one thread per (b,c); loop n ascending (last idx wins)
__global__ void k_scatter_seq(const float* desc, const int* kp, float* gif,
                              int Nk, int Wf, int shift, long HW, long total) {
  GS(i, total) {
    int c = (int)(i % 256); int b = (int)(i / 256);
    const float* d = desc + ((long)b * 256 + c) * Nk;
    float* gp = gif + ((long)b * 256 + c) * HW;
    for (int n = 0; n < Nk; ++n) {
      int r0 = kp[((long)b * Nk + n) * 2] >> shift;
      int c0 = kp[((long)b * Nk + n) * 2 + 1] >> shift;
      gp[(long)r0 * Wf + c0] = d[n];
    }
  }
}

__global__ void k_kenc_in(const int* kp, float* x, int Nk, int Hf, int Wf, int shift, long total) {
  GS(i, total) {
    int n = (int)(i % Nk); int b = (int)(i / Nk);
    float r = (float)(kp[((long)b * Nk + n) * 2] >> shift);
    float c = (float)(kp[((long)b * Nk + n) * 2 + 1] >> shift);
    float den = 1.f / (0.7f * (float)(Hf > Wf ? Hf : Wf));
    float* xb = x + (long)b * 3 * Nk;
    xb[n]          = (r - 0.5f * Hf) * den;
    xb[Nk + n]     = (c - 0.5f * Wf) * den;
    xb[2 * Nk + n] = 0.f;
  }
}

__global__ __launch_bounds__(256) void k_softmax(float* x, int cols) {
  float* row = x + (long)blockIdx.x * cols;
  __shared__ float sh[256];
  int tid = threadIdx.x;
  float mx = -3.4e38f;
  for (int j = tid; j < cols; j += 256) mx = fmaxf(mx, row[j]);
  sh[tid] = mx; __syncthreads();
  for (int st = 128; st > 0; st >>= 1) { if (tid < st) sh[tid] = fmaxf(sh[tid], sh[tid + st]); __syncthreads(); }
  mx = sh[0]; __syncthreads();
  float s = 0.f;
  for (int j = tid; j < cols; j += 256) s += __expf(row[j] - mx);
  sh[tid] = s; __syncthreads();
  for (int st = 128; st > 0; st >>= 1) { if (tid < st) sh[tid] += sh[tid + st]; __syncthreads(); }
  float inv = 1.f / sh[0];
  for (int j = tid; j < cols; j += 256) row[j] = __expf(row[j] - mx) * inv;
}

__global__ void k_build_Z(const float* sc, float* Z, int Nn, float alpha, long total) {
  int M1 = Nn + 1;
  GS(i, total) {
    int j = (int)(i % M1); long r = i / M1;
    int ii = (int)(r % M1); int b = (int)(r / M1);
    Z[i] = (ii < Nn && j < Nn) ? sc[((long)b * Nn + ii) * Nn + j] : alpha;
  }
}

// persistent Sinkhorn: one workgroup per batch, u/v in LDS, 100 dependent sweeps
__global__ __launch_bounds__(256) void k_sinkhorn(const float* __restrict__ Z,
                                                  float* __restrict__ uo, float* __restrict__ vo,
                                                  int Nn, int iters) {
  const int M1 = Nn + 1;
  __shared__ float u[516], v[516];
  const int b = blockIdx.x, tid = threadIdx.x;
  const float* Zb = Z + (long)b * M1 * M1;
  const float norm  = -__logf((float)(2 * Nn));
  const float bonus = __logf((float)Nn) + norm;
  for (int i = tid; i < M1; i += 256) { u[i] = 0.f; v[i] = 0.f; }
  __syncthreads();
  for (int it = 0; it < iters; ++it) {
    for (int r = tid; r < M1; r += 256) {
      const float* row = Zb + (long)r * M1;
      float mx = -3.4e38f;
      for (int j = 0; j < M1; ++j) mx = fmaxf(mx, row[j] + v[j]);
      float s = 0.f;
      for (int j = 0; j < M1; ++j) s += __expf(row[j] + v[j] - mx);
      u[r] = (r < Nn ? norm : bonus) - (mx + __logf(s));
    }
    __syncthreads();
    for (int c = tid; c < M1; c += 256) {
      float mx = -3.4e38f;
      for (int i = 0; i < M1; ++i) mx = fmaxf(mx, Zb[(long)i * M1 + c] + u[i]);
      float s = 0.f;
      for (int i = 0; i < M1; ++i) s += __expf(Zb[(long)i * M1 + c] + u[i] - mx);
      v[c] = (c < Nn ? norm : bonus) - (mx + __logf(s));
    }
    __syncthreads();
  }
  for (int i = tid; i < M1; i += 256) { uo[(long)b * M1 + i] = u[i]; vo[(long)b * M1 + i] = v[i]; }
}

__global__ void k_P(const float* Z, const float* u, const float* v, float* P,
                    int Nn, float norm, long total) {
  int M1 = Nn + 1;
  GS(i, total) {
    int j = (int)(i % Nn); long r = i / Nn;
    int ii = (int)(r % Nn); int b = (int)(r / Nn);
    P[i] = __expf(Z[((long)b * M1 + ii) * M1 + j] + u[(long)b * M1 + ii] + v[(long)b * M1 + j] - norm);
  }
}

// ===================== host orchestration =====================
static inline unsigned cdivu(long a, long b) { return (unsigned)((a + b - 1) / b); }
static inline unsigned g1d(long total) {
  long g = (total + 255) / 256; return (unsigned)(g < 8192 ? g : 8192);
}

struct GnnP { const float *b1,*b2,*bk,*bm,*bq,*bv,*w1,*w2,*wk,*wm,*wq,*wv; };
struct GP {
  const float *d1_b,*d1_w,*d2_b,*d2_w,*fmlp_b,*fmlp_w,*fp_b,*fp_w;
  GnnP gnn[8];
  const float *kenc_w[4], *kenc_b[4];
};

extern "C" void kernel_launch(void* const* d_in, const int* in_sizes, int n_in,
                              void* d_out, int out_size, void* d_ws, size_t ws_size,
                              hipStream_t stream) {
  const float* image = (const float*)d_in[0];
  const int*   kpts  = (const int*)d_in[1];
  const int Bn = in_sizes[0] / (256 * 256);            // 2
  const int Nk = in_sizes[1] / (2 * Bn);               // 512

  // ---- params in JAX pytree order (alphabetical keys) ----
  int cur = 2;
  auto nxt = [&]() { return (const float*)d_in[cur++]; };
  const float *c_w[8], *c_b[8];
  for (int i = 0; i < 8; ++i) { c_w[i] = nxt(); c_b[i] = nxt(); }       // c1_1..c4_2
  const float *cen_w[2], *cen_b[2];
  for (int i = 0; i < 2; ++i) { cen_w[i] = nxt(); cen_b[i] = nxt(); }   // center1,2
  const float *e_w[8], *e_b[8];
  for (int i = 0; i < 8; ++i) { e_w[i] = nxt(); e_b[i] = nxt(); }       // e1_1..e4_2
  static GP G[4];
  for (int l = 0; l < 4; ++l) {                                         // g1..g4
    GP& g = G[l];
    g.d1_b = nxt(); g.d1_w = nxt(); g.d2_b = nxt(); g.d2_w = nxt();
    g.fmlp_b = nxt(); g.fmlp_w = nxt(); g.fp_b = nxt(); g.fp_w = nxt();
    for (int k = 0; k < 8; ++k) {
      GnnP& L = g.gnn[k];
      L.b1 = nxt(); L.b2 = nxt(); L.bk = nxt(); L.bm = nxt(); L.bq = nxt(); L.bv = nxt();
      L.w1 = nxt(); L.w2 = nxt(); L.wk = nxt(); L.wm = nxt(); L.wq = nxt(); L.wv = nxt();
    }
    for (int k = 0; k < 4; ++k) { g.kenc_w[k] = nxt(); g.kenc_b[k] = nxt(); }
  }
  const float* headW = nxt();
  const float *up_w[4], *up_b[4];
  for (int i = 0; i < 4; ++i) { up_w[i] = nxt(); up_b[i] = nxt(); }     // up2..up5

  // ---- workspace bump allocator ----
  size_t off = 0;
  auto alloc = [&](long nFloats) -> float* {
    float* p = (float*)((char*)d_ws + off);
    off += (size_t)(((long)nFloats * 4 + 255) & ~255L);
    return p;
  };
  const int Hs[4] = {256, 128, 64, 32};
  const int C1s[4] = {32, 64, 128, 256};
  float* bufP   = alloc((long)Bn * 64 * 65536);
  float* bufQ   = alloc((long)Bn * 64 * 65536);
  float* bufCat = alloc((long)Bn * 128 * 65536);
  float* bufU   = alloc((long)Bn * 4 * 64 * 16384);
  bf16*  wbf    = (bf16*)alloc(5L * 1024 * 1024);      // 10M bf16 weight scratch
  bf16*  wpbf   = (bf16*)alloc(4L * 512 * 1024 / 2);   // packed up-conv weights
  float* cat[4];
  for (int l = 0; l < 4; ++l) cat[l] = alloc((long)Bn * 2 * C1s[l] * Hs[l] * Hs[l]);
  float* gif   = alloc((long)Bn * 256 * 65536);
  float* dbuf  = alloc((long)Bn * 256 * Nk);
  float* k1    = alloc((long)Bn * 256 * Nk);
  float* k2    = alloc((long)Bn * 256 * Nk);
  float* qb    = alloc((long)Bn * 256 * Nk);
  float* kb    = alloc((long)Bn * 256 * Nk);
  float* vb    = alloc((long)Bn * 256 * Nk);
  float* msgb  = alloc((long)Bn * 256 * Nk);
  float* mdbuf = alloc((long)Bn * 256 * Nk);
  float* catm  = alloc((long)Bn * 512 * Nk);
  float* h1    = alloc((long)Bn * 512 * Nk);
  float* att   = alloc((long)Bn * 4 * Nk * Nk);
  float* scb   = alloc((long)Bn * Nk * Nk);
  float* Zb    = alloc((long)Bn * (Nk + 1) * (Nk + 1));
  float* Pb    = alloc((long)Bn * Nk * Nk);
  float* ub    = alloc((long)Bn * (Nk + 1));
  float* vvb   = alloc((long)Bn * (Nk + 1));
  float* meanB = alloc(1024);
  float* rstdB = alloc(1024);

  float* outSeg    = (float*)d_out;
  float* outSparse = outSeg + (long)Bn * 4 * 65536;

  // ---- launch helpers ----
  auto gemm = [&](const bf16* Ah, const float* Af, long lda, int tA, long sAb, long sAh,
                  const float* B, long ldb, int tB, long sBb, long sBh,
                  float* C, long ldc, long sCb, long sCh,
                  const float* bias, int M, int N, int K,
                  float scale, int act, int accum, int Zc, int zDiv) {
    GemmArgs a;
    a.Ah = Ah; a.Af = Af; a.Bm = B; a.C = C; a.bias = bias;
    a.lda = lda; a.ldb = ldb; a.ldc = ldc;
    a.sAb = sAb; a.sAh = sAh; a.sBb = sBb; a.sBh = sBh; a.sCb = sCb; a.sCh = sCh;
    a.M = M; a.N = N; a.K = K; a.zDiv = zDiv; a.transA = tA; a.transB = tB;
    a.act = act; a.accum = accum; a.scale = scale;
    dim3 grid(cdivu(N, TN), cdivu(M, TM), (unsigned)Zc);
    k_gemm_wmma<<<grid, NTHR, 0, stream>>>(a);
  };
  // cast one weight matrix into the rolling bf16 scratch (stream-ordered reuse)
  auto castW = [&](const float* w, long n) -> const bf16* {
    k_cast<<<g1d(n), 256, 0, stream>>>(w, wbf, n);
    return wbf;
  };
  auto conv3 = [&](const float* Wf, const float* X, float* Y, const float* bias,
                   int Cin, int Cout, int H, long sXz, long sYz, int act) {
    long nW = (long)Cout * Cin * 9;
    k_cast<<<g1d(nW), 256, 0, stream>>>(Wf, wbf, nW);
    ConvArgs a; a.Wt = wbf; a.X = X; a.Y = Y; a.bias = bias;
    a.Cin = Cin; a.Cout = Cout; a.H = H; a.Wd = H; a.sXz = sXz; a.sYz = sYz; a.act = act;
    dim3 grid(cdivu((long)H * H, TN), cdivu(Cout, TM), (unsigned)Bn);
    k_conv3_wmma<<<grid, NTHR, 0, stream>>>(a);
  };
  auto bn = [&](float* x, int C, long chStride, long zStride, int HW, int act) {
    k_bn_stats<<<dim3(C), dim3(256), 0, stream>>>(x, meanB, rstdB, chStride, zStride, Bn, HW);
    long total = (long)Bn * C * HW;
    k_bn_apply<<<g1d(total), 256, 0, stream>>>(x, meanB, rstdB, C, chStride, zStride, HW, act, total);
  };

  // ---- graph layer (gather -> SuperGlue -> Sinkhorn -> scatter -> 2 convs) ----
  auto graph_layer = [&](GP& g, int lvl, float* catL, int C1, int Hf) {
    const int shift = lvl;
    const long HW = (long)Hf * Hf;
    const long catZ = (long)2 * C1 * HW;
    k_gather<<<g1d((long)Bn * C1 * Nk), 256, 0, stream>>>(
        catL, kpts, k2, C1, Nk, Hf, shift, catZ, HW, (long)Bn * C1 * Nk);
    gemm(castW(g.fmlp_w, 256L * C1), nullptr, C1, 0, 0, 0, k2, Nk, 0, (long)C1 * Nk, 0,
         dbuf, Nk, 256L * Nk, 0, g.fmlp_b, 256, Nk, C1, 1.f, 0, 0, Bn, 1);
    k_kenc_in<<<g1d((long)Bn * Nk), 256, 0, stream>>>(kpts, k1, Nk, Hf, Hf, shift, (long)Bn * Nk);
    gemm(castW(g.kenc_w[0], 32L * 3),    nullptr, 3,   0, 0, 0, k1, Nk, 0, 3L * Nk,   0, k2, Nk, 32L * Nk,  0, g.kenc_b[0], 32,  Nk, 3,   1.f, 1, 0, Bn, 1);
    gemm(castW(g.kenc_w[1], 64L * 32),   nullptr, 32,  0, 0, 0, k2, Nk, 0, 32L * Nk,  0, k1, Nk, 64L * Nk,  0, g.kenc_b[1], 64,  Nk, 32,  1.f, 1, 0, Bn, 1);
    gemm(castW(g.kenc_w[2], 128L * 64),  nullptr, 64,  0, 0, 0, k1, Nk, 0, 64L * Nk,  0, k2, Nk, 128L * Nk, 0, g.kenc_b[2], 128, Nk, 64,  1.f, 1, 0, Bn, 1);
    gemm(castW(g.kenc_w[3], 256L * 128), nullptr, 128, 0, 0, 0, k2, Nk, 0, 128L * Nk, 0, k1, Nk, 256L * Nk, 0, g.kenc_b[3], 256, Nk, 128, 1.f, 0, 0, Bn, 1);
    k_add<<<g1d((long)Bn * 256 * Nk), 256, 0, stream>>>(dbuf, dbuf, k1, (long)Bn * 256 * Nk);
    const long sD = 256L * Nk, sC2 = 512L * Nk, sNN = (long)Nk * Nk;
    for (int L = 0; L < 8; ++L) {
      GnnP& p = g.gnn[L];
      gemm(castW(p.wq, 256L * 256), nullptr, 256, 0, 0, 0, dbuf, Nk, 0, sD, 0, qb, Nk, sD, 0, p.bq, 256, Nk, 256, 1.f, 0, 0, Bn, 1);
      gemm(castW(p.wk, 256L * 256), nullptr, 256, 0, 0, 0, dbuf, Nk, 0, sD, 0, kb, Nk, sD, 0, p.bk, 256, Nk, 256, 1.f, 0, 0, Bn, 1);
      gemm(castW(p.wv, 256L * 256), nullptr, 256, 0, 0, 0, dbuf, Nk, 0, sD, 0, vb, Nk, sD, 0, p.bv, 256, Nk, 256, 1.f, 0, 0, Bn, 1);
      // scores[b,h] = q_h^T k_h / 8  (f32 A, transA; head views lda = 4*Nk)
      gemm(nullptr, qb, 4L * Nk, 1, sD, Nk, kb, 4L * Nk, 0, sD, Nk,
           att, Nk, 4L * sNN, sNN, nullptr, Nk, Nk, 64, 0.125f, 0, 0, Bn * 4, 4);
      k_softmax<<<dim3((unsigned)(Bn * 4 * Nk)), 256, 0, stream>>>(att, Nk);
      // msg_h = v_h * att^T  (f32 A)
      gemm(nullptr, vb, 4L * Nk, 0, sD, Nk, att, Nk, 1, 4L * sNN, sNN,
           msgb, 4L * Nk, sD, Nk, nullptr, 64, Nk, Nk, 1.f, 0, 0, Bn * 4, 4);
      k_copy_ch<<<g1d((long)Bn * 256 * Nk), 256, 0, stream>>>(dbuf, catm, 256, Nk, sD, sC2, (long)Bn * 256 * Nk);
      gemm(castW(p.wm, 256L * 256), nullptr, 256, 0, 0, 0, msgb, Nk, 0, sD, 0, catm + 256L * Nk, Nk, sC2, 0,
           p.bm, 256, Nk, 256, 1.f, 0, 0, Bn, 1);
      gemm(castW(p.w1, 512L * 512), nullptr, 512, 0, 0, 0, catm, Nk, 0, sC2, 0, h1, Nk, sC2, 0, p.b1, 512, Nk, 512, 1.f, 1, 0, Bn, 1);
      gemm(castW(p.w2, 256L * 512), nullptr, 512, 0, 0, 0, h1, Nk, 0, sC2, 0, dbuf, Nk, sD, 0, p.b2, 256, Nk, 512, 1.f, 0, 1, Bn, 1);
    }
    gemm(castW(g.fp_w, 256L * 256), nullptr, 256, 0, 0, 0, dbuf, Nk, 0, sD, 0, mdbuf, Nk, sD, 0, g.fp_b, 256, Nk, 256, 1.f, 0, 0, Bn, 1);
    gemm(nullptr, mdbuf, Nk, 1, sD, 0, mdbuf, Nk, 0, sD, 0, scb, Nk, sNN, 0,
         nullptr, Nk, Nk, 256, 0.25f, 0, 0, Bn, 1);   // 1/256^0.25
    const int M1 = Nk + 1;
    k_build_Z<<<g1d((long)Bn * M1 * M1), 256, 0, stream>>>(scb, Zb, Nk, 1.0f, (long)Bn * M1 * M1);
    k_sinkhorn<<<dim3((unsigned)Bn), 256, 0, stream>>>(Zb, ub, vvb, Nk, 100);
    k_P<<<g1d((long)Bn * sNN), 256, 0, stream>>>(Zb, ub, vvb, Pb, Nk, -logf((float)(2 * Nk)), (long)Bn * sNN);
    gemm(nullptr, mdbuf, Nk, 0, sD, 0, Pb, Nk, 1, sNN, 0, dbuf, Nk, sD, 0,
         nullptr, 256, Nk, Nk, 1.f, 0, 1, Bn, 1);
    k_copy_ch<<<g1d((long)Bn * 256 * Nk), 256, 0, stream>>>(
        dbuf, outSparse + (long)lvl * 256 * Nk, 256, Nk, sD, 1024L * Nk, (long)Bn * 256 * Nk);
    k_zero<<<g1d((long)Bn * 256 * HW), 256, 0, stream>>>(gif, (long)Bn * 256 * HW);
    k_scatter_seq<<<g1d((long)Bn * 256), 256, 0, stream>>>(dbuf, kpts, gif, Nk, Hf, shift, HW, (long)Bn * 256);
    conv3(g.d1_w, gif, bufQ, g.d1_b, 256, 64, Hf, 256L * HW, 64L * HW, 0);
    bn(bufQ, 64, HW, 64L * HW, (int)HW, 1);
    conv3(g.d2_w, bufQ, catL + (long)C1 * HW, g.d2_b, 64, C1, Hf, 64L * HW, catZ, 0);
  };

  // ================= forward =================
  for (int l = 0; l < 4; ++l) {
    const int H = Hs[l], C1 = C1s[l];
    const long HW = (long)H * H;
    const int Cin = (l == 0) ? 1 : 2 * C1s[l - 1];
    const float* in = (l == 0) ? image : bufP;
    conv3(c_w[2 * l], in, bufQ, c_b[2 * l], Cin, C1, H, (long)Cin * HW, (long)C1 * HW, 0);
    bn(bufQ, C1, HW, (long)C1 * HW, (int)HW, 2);
    conv3(c_w[2 * l + 1], bufQ, cat[l], c_b[2 * l + 1], C1, C1, H, (long)C1 * HW, 2L * C1 * HW, 0);
    bn(cat[l], C1, HW, 2L * C1 * HW, (int)HW, 2);
    graph_layer(G[l], l, cat[l], C1, H);
    const int Ho = H / 2;
    k_pool2<<<g1d((long)Bn * 2 * C1 * Ho * Ho), 256, 0, stream>>>(
        cat[l], bufP, 2 * C1, H, H, (long)Bn * 2 * C1 * Ho * Ho);
  }
  conv3(cen_w[0], bufP, bufQ, cen_b[0], 512, 1024, 16, 512L * 256, 1024L * 256, 1);
  conv3(cen_w[1], bufQ, bufP, cen_b[1], 1024, 1024, 16, 1024L * 256, 1024L * 256, 1);
  const int CinU[4] = {128, 256, 512, 1024}, CoutU[4] = {64, 128, 256, 512};
  for (int dl = 3; dl >= 0; --dl) {
    const int Hin = Hs[dl] / 2, Hout = Hs[dl];
    const long HWin = (long)Hin * Hin, HWout = (long)Hout * Hout;
    const int Ci = CinU[dl], Co = CoutU[dl], Cskip = 2 * C1s[dl], Ccat = Co + Cskip;
    k_pack_upw<<<g1d(4L * Co * Ci), 256, 0, stream>>>(up_w[dl], wpbf, Ci, Co, 4L * Co * Ci);
    for (int sub = 0; sub < 4; ++sub)
      gemm(wpbf + (long)sub * Co * Ci, nullptr, Ci, 0, 0, 0, bufP, HWin, 0, (long)Ci * HWin, 0,
           bufU + (long)sub * Co * HWin, HWin, 4L * Co * HWin, 0,
           nullptr, Co, (int)HWin, Ci, 1.f, 0, 0, Bn, 1);
    k_up_inter<<<g1d((long)Bn * 4 * Co * HWin), 256, 0, stream>>>(
        bufU, up_b[dl], bufCat, Co, Hin, Hin, 4L * Co * HWin, (long)Ccat * HWout,
        (long)Bn * 4 * Co * HWin);
    k_copy_ch<<<g1d((long)Bn * Cskip * HWout), 256, 0, stream>>>(
        cat[dl], bufCat + (long)Co * HWout, Cskip, HWout,
        (long)Cskip * HWout, (long)Ccat * HWout, (long)Bn * Cskip * HWout);
    conv3(e_w[2 * dl], bufCat, bufQ, e_b[2 * dl], Ccat, Co, Hout, (long)Ccat * HWout, (long)Co * HWout, 0);
    bn(bufQ, Co, HWout, (long)Co * HWout, (int)HWout, 2);
    conv3(e_w[2 * dl + 1], bufQ, bufP, e_b[2 * dl + 1], Co, Co, Hout, (long)Co * HWout, (long)Co * HWout, 0);
    bn(bufP, Co, HWout, (long)Co * HWout, (int)HWout, 2);
  }
  gemm(castW(headW, 4L * 64), nullptr, 64, 0, 0, 0, bufP, 65536, 0, 64L * 65536, 0,
       outSeg, 65536, 4L * 65536, 0, nullptr, 4, 65536, 64, 1.f, 0, 0, Bn, 1);
}
